// SpeFEM_23192823398810
// MI455X (gfx1250) — matrix operand order repaired and checked by reading, not compiled, not run on hardware
//
#include <hip/hip_runtime.h>
#include <math.h>

// ---------------- problem constants ----------------
#define BSZ   32
#define CHN   200
#define HWSZ  256          // H*W = 16*16
#define SEQL  200          // sequence length (== CHN)
#define EMB   256          // d_model
#define DI    512          // d_inner
#define NST   16           // d_state
#define DTR   16           // dt_rank
#define NTOK  (BSZ * SEQL)         // 6400 rows for token GEMMs
#define TOTE  (BSZ * CHN * HWSZ)   // 1,638,400 elements of [B,C,H,W]
#define EWTOT ((long long)NTOK * DI)

typedef __attribute__((ext_vector_type(16))) _Float16 v16h;
typedef __attribute__((ext_vector_type(8)))  float    v8f;

// epilogues
#define EPI_NONE     0
#define EPI_LEAKY    1
#define EPI_SIGMOID  2
#define EPI_SOFTPLUS 3
#define EPI_ADDRES   4

template <int EPI>
__device__ __forceinline__ void gemm_epilogue(
    float* __restrict__ Cp, v8f acc, int m0, int hi, int n0, int lml,
    int M, int N, int ldc, int nS, const float* __restrict__ bias,
    const float* __restrict__ Rp, int ldres)
{
  const int ncol = n0 + lml;
  float vout[8];
#pragma unroll
  for (int r = 0; r < 8; ++r) {
    const int m = m0 + r + hi * 8;          // D: M = m0 + r + hi*8
    float v = acc[r];
    if (EPI == EPI_LEAKY)    v = v >= 0.f ? v : 0.2f * v;
    if (EPI == EPI_SIGMOID)  v = 1.f / (1.f + __expf(-v));
    if (EPI == EPI_SOFTPLUS) { v += bias[nS]; v = (v > 20.f) ? v : log1pf(__expf(v)); }
    if (EPI == EPI_ADDRES)   v += Rp[(long long)(m < M ? m : M - 1) * ldres + nS];
    vout[r] = v;
  }
  if ((m0 + 16 <= M) && (n0 + 16 <= N)) {   // wave-uniform interior test
#pragma unroll
    for (int r = 0; r < 8; ++r)
      Cp[(long long)(m0 + r + hi * 8) * ldc + ncol] = vout[r];
  } else {
#pragma unroll
    for (int r = 0; r < 8; ++r) {
      const int m = m0 + r + hi * 8;
      if (m < M && ncol < N) Cp[(long long)m * ldc + ncol] = vout[r];
    }
  }
}

// =====================================================================
// WMMA GEMM: C[z] = A * B[z] (+epilogue). A: MxK f32 row-major (lda).
// B: KxN f32 row-major (compile-time LDB, batch stride sB). C: MxN (ldc).
// One wave handles TWO adjacent 16x16 N-tiles (A fragment reused across
// two back-to-back WMMAs); 4 waves per block -> 8 N-tiles per block.
//
// Out-of-range handling without exec divergence:
//  - A rows / B cols are CLAMPED (garbage only reaches D rows/cols that are
//    never stored).
//  - K edge is a wave-uniform branch; partial step uses clamped addresses
//    plus value selects (v_cndmask), still branch-free per lane.
// LDB is a template constant so the 16 strided B loads per fragment become
// independent global_load_b32 with immediate offsets (clauseable, one wait)
// instead of a serial v_add_u64 / s_wait_xcnt chain.
// =====================================================================
template <int EPI, int LDB>
__global__ __launch_bounds__(128) void gemm_wmma(
    const float* __restrict__ A, const float* __restrict__ B,
    float* __restrict__ C, int M, int N, int K, int lda, int ldc,
    long long sB, long long sC, const float* __restrict__ bias,
    const float* __restrict__ res, int ldres, long long sRes)
{
  const int lane = threadIdx.x & 31;
  const int wid  = threadIdx.x >> 5;
  const int tm   = blockIdx.x;
  const int tn   = (blockIdx.y * 4 + wid) * 2;     // two N tiles per wave
  if (tn * 16 >= N) return;                        // whole wave exits together
  const int z = blockIdx.z;
  const float* Bp = B + (long long)z * sB;
  float*       Cp = C + (long long)z * sC;

  const int m0  = tm * 16;
  const int n0  = tn * 16, n1 = n0 + 16;
  const int lml = lane & 15;
  const int hi  = lane >> 4;                       // 0 or 1 (lane group)
  const int mrow = m0 + lml;
  const int mS  = (mrow < M) ? mrow : (M - 1);     // clamped (safe, see above)
  const int nc0 = n0 + lml, nc1 = n1 + lml;
  const int nS0 = (nc0 < N) ? nc0 : (N - 1);
  const int nS1 = (nc1 < N) ? nc1 : (N - 1);

  const float* Arow = A + (long long)mS * lda;
  const float* Bc0  = Bp + nS0;
  const float* Bc1  = Bp + nS1;

  v8f acc0 = {}, acc1 = {};
  for (int k0 = 0; k0 < K; k0 += 32) {
    v16h a, b0, b1;
    const int kbA = k0 + hi * 16;                  // A: K = kbA + e
    const int kbB = k0 + hi * 16;                  // B: K = kbB + e
    if (k0 + 32 <= K) {
      // ---- fast path: straight-line vector loads, no predication ----
      const float4 a0v = *(const float4*)(Arow + kbA);
      const float4 a1v = *(const float4*)(Arow + kbA + 4);
      const float4 a2v = *(const float4*)(Arow + kbA + 8);
      const float4 a3v = *(const float4*)(Arow + kbA + 12);
      a[0] = (_Float16)a0v.x;  a[1] = (_Float16)a0v.y;
      a[2] = (_Float16)a0v.z;  a[3] = (_Float16)a0v.w;
      a[4] = (_Float16)a1v.x;  a[5] = (_Float16)a1v.y;
      a[6] = (_Float16)a1v.z;  a[7] = (_Float16)a1v.w;
      a[8] = (_Float16)a2v.x;  a[9] = (_Float16)a2v.y;
      a[10] = (_Float16)a2v.z; a[11] = (_Float16)a2v.w;
      a[12] = (_Float16)a3v.x; a[13] = (_Float16)a3v.y;
      a[14] = (_Float16)a3v.z; a[15] = (_Float16)a3v.w;
      const float* p0 = Bc0 + (long long)kbB * LDB;
      const float* p1 = Bc1 + (long long)kbB * LDB;
#pragma unroll
      for (int e = 0; e < 16; ++e) b0[e] = (_Float16)p0[e * LDB];
#pragma unroll
      for (int e = 0; e < 16; ++e) b1[e] = (_Float16)p1[e * LDB];
    } else {
      // ---- K tail: clamped addresses + zero-select, branch-free ----
#pragma unroll
      for (int e = 0; e < 16; ++e) {
        const int kk = kbA + e;
        const float v = Arow[kk < K ? kk : 0];
        a[e] = (_Float16)(kk < K ? v : 0.f);
      }
#pragma unroll
      for (int e = 0; e < 16; ++e) {
        const int kk = kbB + e;
        const int kc = kk < K ? kk : 0;
        const float v0 = Bc0[(long long)kc * LDB];
        const float v1 = Bc1[(long long)kc * LDB];
        b0[e] = (_Float16)(kk < K ? v0 : 0.f);
        b1[e] = (_Float16)(kk < K ? v1 : 0.f);
      }
    }
    acc0 = __builtin_amdgcn_wmma_f32_16x16x32_f16(false, a, false, b0,
                                                  (short)0, acc0, false, false);
    acc1 = __builtin_amdgcn_wmma_f32_16x16x32_f16(false, a, false, b1,
                                                  (short)0, acc1, false, false);
  }

  const float* Rp = (EPI == EPI_ADDRES) ? (res + (long long)z * sRes) : nullptr;
  gemm_epilogue<EPI>(Cp, acc0, m0, hi, n0, lml, M, N, ldc, nS0, bias, Rp, ldres);
  gemm_epilogue<EPI>(Cp, acc1, m0, hi, n1, lml, M, N, ldc, nS1, bias, Rp, ldres);
}

// ---------------- BatchNorm (stats over B,H,W per channel) ----------------
__global__ void bn_stats(const float* __restrict__ x, const float* __restrict__ g,
                         const float* __restrict__ b, float* __restrict__ scale,
                         float* __restrict__ shift)
{
  __shared__ float s1[256], s2[256];
  const int c = blockIdx.x, t = threadIdx.x;
  float a1 = 0.f, a2 = 0.f;
  for (int i = t; i < BSZ * HWSZ; i += 256) {
    const int bb = i >> 8, hw = i & 255;
    const float v = x[((long long)(bb * CHN + c)) * HWSZ + hw];
    a1 += v; a2 += v * v;
  }
  s1[t] = a1; s2[t] = a2; __syncthreads();
  for (int s = 128; s > 0; s >>= 1) {
    if (t < s) { s1[t] += s1[t + s]; s2[t] += s2[t + s]; }
    __syncthreads();
  }
  if (t == 0) {
    const float mean = s1[0] / (float)(BSZ * HWSZ);
    const float var  = s2[0] / (float)(BSZ * HWSZ) - mean * mean;
    const float sc   = g[c] * rsqrtf(var + 1e-5f);
    scale[c] = sc; shift[c] = b[c] - mean * sc;
  }
}

__global__ void bn_apply(const float* __restrict__ x, const float* __restrict__ scale,
                         const float* __restrict__ shift, float* __restrict__ out)
{
  const long long i = (long long)blockIdx.x * blockDim.x + threadIdx.x;
  if (i >= TOTE) return;
  const int c = (int)((i >> 8) % CHN);
  out[i] = x[i] * scale[c] + shift[c];
}

// ---------------- spatial attention + x1 = xn*attn ----------------
__global__ void spatial_attn(const float* __restrict__ xn, const float* __restrict__ w,
                             float* __restrict__ x1)
{
  __shared__ float avg[256], mxs[256];
  const int b = blockIdx.x, t = threadIdx.x;     // t = hw
  float a = 0.f, m = -1e30f;
  for (int c = 0; c < CHN; ++c) {
    const float v = xn[((long long)(b * CHN + c)) * HWSZ + t];
    a += v; m = fmaxf(m, v);
  }
  avg[t] = a / (float)CHN; mxs[t] = m;
  __syncthreads();
  const int y = t >> 4, x = t & 15;
  float acc = 0.f;
  for (int ky = 0; ky < 7; ++ky)
    for (int kx = 0; kx < 7; ++kx) {
      const int yy = y + ky - 3, xx = x + kx - 3;
      if (yy >= 0 && yy < 16 && xx >= 0 && xx < 16) {
        acc += w[ky * 7 + kx] * avg[yy * 16 + xx];
        acc += w[49 + ky * 7 + kx] * mxs[yy * 16 + xx];
      }
    }
  const float attn = 1.f / (1.f + __expf(-acc));
  for (int c = 0; c < CHN; ++c) {
    const long long i = ((long long)(b * CHN + c)) * HWSZ + t;
    x1[i] = xn[i] * attn;
  }
}

// ---------------- LayerNorm over EMB=256 ----------------
__global__ void layernorm(const float* __restrict__ x, const float* __restrict__ g,
                          const float* __restrict__ b, float* __restrict__ out)
{
  __shared__ float s1[256], s2[256];
  const int row = blockIdx.x, t = threadIdx.x;
  const float v = x[(long long)row * EMB + t];
  s1[t] = v; s2[t] = v * v; __syncthreads();
  for (int s = 128; s > 0; s >>= 1) {
    if (t < s) { s1[t] += s1[t + s]; s2[t] += s2[t + s]; }
    __syncthreads();
  }
  const float mu  = s1[0] / (float)EMB;
  const float var = s2[0] / (float)EMB - mu * mu;
  out[(long long)row * EMB + t] = (v - mu) * rsqrtf(var + 1e-5f) * g[t] + b[t];
}

// ---------------- causal depthwise conv (K=3) + SiLU; optional flip ----------------
__global__ void dwconv_silu(const float* __restrict__ xin, const float* __restrict__ w,
                            const float* __restrict__ bias, float* __restrict__ u, int rev)
{
  const long long idx = (long long)blockIdx.x * blockDim.x + threadIdx.x;
  if (idx >= EWTOT) return;
  const int d = (int)(idx & (DI - 1));
  const int l = (int)((idx >> 9) % SEQL);
  const int b = (int)(idx / ((long long)DI * SEQL));
  float acc = bias[d];
#pragma unroll
  for (int k = 0; k < 3; ++k) {
    const int lt = l + k - 2;
    if (lt >= 0) {
      const int lsrc = rev ? (SEQL - 1 - lt) : lt;
      acc += xin[((long long)(b * SEQL + lsrc)) * 1024 + d] * w[d * 3 + k];
    }
  }
  u[((long long)(b * SEQL + l)) * DI + d] = acc / (1.f + __expf(-acc));
}

// ---------------- selective scan: one thread per (b, d_inner) ----------------
__global__ void sel_scan(const float* __restrict__ u, const float* __restrict__ dt,
                         const float* __restrict__ xdbl, const float* __restrict__ A_log,
                         const float* __restrict__ Dp, float* __restrict__ y)
{
  const int idx = blockIdx.x * blockDim.x + threadIdx.x;
  if (idx >= BSZ * DI) return;
  const int d = idx & (DI - 1);
  const int b = idx >> 9;
  float A[NST];
#pragma unroll
  for (int n = 0; n < NST; ++n) A[n] = -__expf(A_log[d * NST + n]);
  const float Dv = Dp[d];
  float h[NST];
#pragma unroll
  for (int n = 0; n < NST; ++n) h[n] = 0.f;
  for (int l = 0; l < SEQL; ++l) {
    const long long t  = (long long)(b * SEQL + l);
    const float ut  = u[t * DI + d];
    const float dtt = dt[t * DI + d];
    const float* Bc = xdbl + t * 48 + DTR;
    const float* Cc = xdbl + t * 48 + DTR + NST;
    const float du = dtt * ut;
    float yv = 0.f;
#pragma unroll
    for (int n = 0; n < NST; ++n) {
      h[n] = __expf(dtt * A[n]) * h[n] + du * Bc[n];
      yv += h[n] * Cc[n];
    }
    y[t * DI + d] = yv + ut * Dv;
  }
}

// ---------------- merge forward/backward + gate with silu(z) ----------------
__global__ void merge_silu(const float* __restrict__ yf, const float* __restrict__ yb,
                           const float* __restrict__ xin, const float* __restrict__ mergep,
                           float* __restrict__ y)
{
  const long long idx = (long long)blockIdx.x * blockDim.x + threadIdx.x;
  if (idx >= EWTOT) return;
  const int d = (int)(idx & (DI - 1));
  const int l = (int)((idx >> 9) % SEQL);
  const int b = (int)(idx / ((long long)DI * SEQL));
  const float m0 = mergep[0], m1 = mergep[1];
  const float mx = fmaxf(m0, m1);
  const float e0 = __expf(m0 - mx), e1 = __expf(m1 - mx);
  const float w0 = e0 / (e0 + e1), w1 = e1 / (e0 + e1);
  const long long t  = (long long)(b * SEQL + l);
  const long long tr = (long long)(b * SEQL + (SEQL - 1 - l));
  const float z  = xin[t * 1024 + DI + d];
  const float sz = z / (1.f + __expf(-z));
  y[t * DI + d] = (w0 * yf[t * DI + d] + w1 * yb[tr * DI + d]) * sz;
}

// ---------------- direct 3x3 conv (SCPA branches), optional leaky ----------------
__global__ void conv3x3(const float* __restrict__ in, const float* __restrict__ w,
                        float* __restrict__ out, int Cin, int Cout,
                        long long inBS, long long outBS, int leaky)
{
  const int idx = blockIdx.x * blockDim.x + threadIdx.x;
  const int total = BSZ * Cout * HWSZ;
  if (idx >= total) return;
  const int hw = idx & 255;
  const int o  = (idx >> 8) % Cout;
  const int b  = idx / (HWSZ * Cout);
  const int y = hw >> 4, x = hw & 15;
  const float* ib = in + (long long)b * inBS;
  float acc = 0.f;
  for (int i = 0; i < Cin; ++i) {
    const float* ip = ib + (long long)i * HWSZ;
    const float* wp = w + ((long long)o * Cin + i) * 9;
#pragma unroll
    for (int ky = 0; ky < 3; ++ky) {
      const int yy = y + ky - 1;
      if (yy < 0 || yy > 15) continue;
#pragma unroll
      for (int kx = 0; kx < 3; ++kx) {
        const int xx = x + kx - 1;
        if (xx < 0 || xx > 15) continue;
        acc += wp[ky * 3 + kx] * ip[yy * 16 + xx];
      }
    }
  }
  if (leaky) acc = acc >= 0.f ? acc : 0.2f * acc;
  out[(long long)b * outBS + (long long)o * HWSZ + hw] = acc;
}

// ---------------- elementwise helpers ----------------
__global__ void add3_ew(const float* __restrict__ a, const float* __restrict__ b,
                        const float* __restrict__ c, float* __restrict__ out, long long n)
{
  const long long i = (long long)blockIdx.x * blockDim.x + threadIdx.x;
  if (i < n) out[i] = a[i] + b[i] + c[i];
}
__global__ void mul_ew(const float* __restrict__ a, const float* __restrict__ b,
                       float* __restrict__ out, long long n)
{
  const long long i = (long long)blockIdx.x * blockDim.x + threadIdx.x;
  if (i < n) out[i] = a[i] * b[i];
}

// =====================================================================
// Host side
// =====================================================================
static dim3 gemm_grid(int M, int N, int batch)
{
  const int ntiles = (N + 15) / 16;
  return dim3((M + 15) / 16, (ntiles + 7) / 8, batch);  // 8 N-tiles per block
}

#define GEMM_LAUNCH(EPI_, LDB_)                                                   \
  gemm_wmma<EPI_, LDB_><<<gemm_grid(M, N, batch), dim3(128), 0, s>>>(             \
      A, B, C, M, N, K, lda, ldc, sB, sC, bias, res, ldres, sRes)

static void launch_gemm(hipStream_t s, int epi, const float* A, const float* B, float* C,
                        int M, int N, int K, int lda, int ldb, int ldc,
                        long long sB, long long sC, int batch,
                        const float* bias, const float* res, int ldres, long long sRes)
{
  // only the (epi, ldb) combinations this model actually uses
  if (ldb == 1024)      { GEMM_LAUNCH(EPI_NONE, 1024); }
  else if (ldb == 48)   { GEMM_LAUNCH(EPI_NONE, 48); }
  else if (ldb == 512)  { GEMM_LAUNCH(EPI_SOFTPLUS, 512); }
  else { // ldb == 256
    switch (epi) {
      case EPI_NONE:    GEMM_LAUNCH(EPI_NONE, 256);    break;
      case EPI_LEAKY:   GEMM_LAUNCH(EPI_LEAKY, 256);   break;
      case EPI_SIGMOID: GEMM_LAUNCH(EPI_SIGMOID, 256); break;
      case EPI_ADDRES:  GEMM_LAUNCH(EPI_ADDRES, 256);  break;
    }
  }
}

extern "C" void kernel_launch(void* const* d_in, const int* in_sizes, int n_in,
                              void* d_out, int out_size, void* d_ws, size_t ws_size,
                              hipStream_t stream)
{
  (void)in_sizes; (void)n_in; (void)out_size; (void)ws_size;
  // ---- inputs (setup_inputs dict insertion order, flattened depth-first) ----
  const float* x     = (const float*)d_in[0];
  const float* bn1_g = (const float*)d_in[1];
  const float* bn1_b = (const float*)d_in[2];
  const float* bn2_g = (const float*)d_in[3];
  const float* bn2_b = (const float*)d_in[4];
  const float* sa_w  = (const float*)d_in[5];
  const float* c1a   = (const float*)d_in[6];
  const float* c1b   = (const float*)d_in[7];
  const float* k1    = (const float*)d_in[8];
  const float* pak2  = (const float*)d_in[9];
  const float* pak3  = (const float*)d_in[10];
  const float* pak4  = (const float*)d_in[11];
  const float* c3w   = (const float*)d_in[12];
  auto blkp = [&](int blk, int j) -> const float* {
    return (const float*)d_in[13 + blk * 19 + j];
  };
  // per-block offsets: 0 ln_g, 1 ln_b, 2 w_in, 3 w_out, 4 merge,
  // then per direction base 5+dir*7: +0 conv_w, +1 conv_b, +2 w_xproj,
  // +3 w_dt, +4 b_dt, +5 A_log, +6 D

  // ---- workspace bump allocator ----
  char* base = (char*)d_ws;
  size_t off = 0;
  auto alloc = [&](size_t elems) -> float* {
    float* r = (float*)(base + off);
    off += ((elems * sizeof(float) + 255) & ~(size_t)255);
    return r;
  };
  float* bn_scale = alloc(CHN);
  float* bn_shift = alloc(CHN);
  float* xnbuf = alloc((size_t)NTOK * EMB);   // BN1 out; doubles as x2 ping
  float* x1buf = alloc((size_t)NTOK * EMB);
  float* x2B   = alloc((size_t)NTOK * EMB);   // x2 pong
  float* xnln  = alloc((size_t)NTOK * EMB);
  float* xin   = alloc((size_t)NTOK * 1024);  // [xh | z]
  float* ubuf  = alloc((size_t)NTOK * DI);
  float* xdbl  = alloc((size_t)NTOK * 48);
  float* dtbuf = alloc((size_t)NTOK * DI);
  float* yf    = alloc((size_t)NTOK * DI);
  float* yb    = alloc((size_t)NTOK * DI);
  // SCPA-phase aliases (block-phase buffers are dead by then)
  const long long PB = (long long)100 * HWSZ;     // per-batch branch plane
  float* a1buf = xin;
  float* b1buf = xin + (size_t)BSZ * PB;
  float* sbuf  = xin + 2 * (size_t)BSZ * PB;
  float* tbuf  = xin + 3 * (size_t)BSZ * PB;
  float* catb  = ubuf;                            // [B, 200, 256]
  float* c3out = dtbuf;
  float* xmid  = yf;
  float* xn2   = yb;

  const int TB = 256;
  const int gTOT = (TOTE + TB - 1) / TB;
  const int gEW  = (int)((EWTOT + TB - 1) / TB);

  // ---- stage 1: BN1, spatial attention ----
  bn_stats<<<CHN, TB, 0, stream>>>(x, bn1_g, bn1_b, bn_scale, bn_shift);
  bn_apply<<<gTOT, TB, 0, stream>>>(x, bn_scale, bn_shift, xnbuf);
  spatial_attn<<<BSZ, TB, 0, stream>>>(xnbuf, sa_w, x1buf);

  // ---- stage 2: 8 SS1D Mamba blocks on [B, L=200, E=256] ----
  float* cur = xnbuf;   // x2 sequence (initially xn)
  float* nxt = x2B;
  for (int blk = 0; blk < 8; ++blk) {
    layernorm<<<NTOK, TB, 0, stream>>>(cur, blkp(blk, 0), blkp(blk, 1), xnln);
    // [6400,256] @ [256,1024] -> [xh | z]
    launch_gemm(stream, EPI_NONE, xnln, blkp(blk, 2), xin,
                NTOK, 2 * DI, EMB, EMB, 2 * DI, 2 * DI, 0, 0, 1,
                nullptr, nullptr, 0, 0);
    for (int dir = 0; dir < 2; ++dir) {
      const float* cw   = blkp(blk, 5 + dir * 7 + 0);
      const float* cb   = blkp(blk, 5 + dir * 7 + 1);
      const float* wxp  = blkp(blk, 5 + dir * 7 + 2);
      const float* wdt  = blkp(blk, 5 + dir * 7 + 3);
      const float* bdt  = blkp(blk, 5 + dir * 7 + 4);
      const float* alog = blkp(blk, 5 + dir * 7 + 5);
      const float* Dp   = blkp(blk, 5 + dir * 7 + 6);
      dwconv_silu<<<gEW, TB, 0, stream>>>(xin, cw, cb, ubuf, dir);
      // u [6400,512] @ w_xproj [512,48]
      launch_gemm(stream, EPI_NONE, ubuf, wxp, xdbl,
                  NTOK, DTR + 2 * NST, DI, DI, 48, 48, 0, 0, 1,
                  nullptr, nullptr, 0, 0);
      // dt = softplus(dt_r [6400,16] @ w_dt [16,512] + b_dt)
      launch_gemm(stream, EPI_SOFTPLUS, xdbl, wdt, dtbuf,
                  NTOK, DI, DTR, 48, DI, DI, 0, 0, 1,
                  bdt, nullptr, 0, 0);
      sel_scan<<<(BSZ * DI + TB - 1) / TB, TB, 0, stream>>>(
          ubuf, dtbuf, xdbl, alog, Dp, dir ? yb : yf);
    }
    merge_silu<<<gEW, TB, 0, stream>>>(yf, yb, xin, blkp(blk, 4), ubuf);
    // x2_next = x2_cur + y [6400,512] @ w_out [512,256]
    launch_gemm(stream, EPI_ADDRES, ubuf, blkp(blk, 3), nxt,
                NTOK, EMB, DI, DI, EMB, EMB, 0, 0, 1,
                nullptr, cur, EMB, 0);
    float* t = cur; cur = nxt; nxt = t;
  }

  // ---- stage 3: x = residual + x1 + x2; BN2 ----
  add3_ew<<<gTOT, TB, 0, stream>>>(x, x1buf, cur, xmid, TOTE);
  bn_stats<<<CHN, TB, 0, stream>>>(xmid, bn2_g, bn2_b, bn_scale, bn_shift);
  bn_apply<<<gTOT, TB, 0, stream>>>(xmid, bn_scale, bn_shift, xn2);

  // ---- stage 4: SCPA ----
  const long long sX = (long long)CHN * HWSZ;   // xn2 batch stride
  // a = leaky(conv1x1_c1a(xn2)); b = leaky(conv1x1_c1b(xn2))  (batched GEMM)
  launch_gemm(stream, EPI_LEAKY, c1a, xn2, a1buf, 100, HWSZ, CHN, CHN, HWSZ, HWSZ,
              sX, PB, BSZ, nullptr, nullptr, 0, 0);
  launch_gemm(stream, EPI_LEAKY, c1b, xn2, b1buf, 100, HWSZ, CHN, CHN, HWSZ, HWSZ,
              sX, PB, BSZ, nullptr, nullptr, 0, 0);
  // a = leaky(conv3x3_k1(a)) -> cat[:, 0:100, :]
  conv3x3<<<(BSZ * 100 * HWSZ + TB - 1) / TB, TB, 0, stream>>>(
      a1buf, k1, catb, 100, 100, PB, 2 * PB, 1);
  // pa = sigmoid(conv1x1_k2(b)) * conv3x3_k3(b)
  launch_gemm(stream, EPI_SIGMOID, pak2, b1buf, sbuf, 100, HWSZ, 100, 100, HWSZ, HWSZ,
              PB, PB, BSZ, nullptr, nullptr, 0, 0);
  conv3x3<<<(BSZ * 100 * HWSZ + TB - 1) / TB, TB, 0, stream>>>(
      b1buf, pak3, tbuf, 100, 100, PB, PB, 0);
  mul_ew<<<(int)(((long long)BSZ * PB + TB - 1) / TB), TB, 0, stream>>>(
      sbuf, tbuf, sbuf, (long long)BSZ * PB);
  // b = leaky(conv3x3_k4(pa)) -> cat[:, 100:200, :]
  conv3x3<<<(BSZ * 100 * HWSZ + TB - 1) / TB, TB, 0, stream>>>(
      sbuf, pak4, catb + 100 * HWSZ, 100, 100, PB, 2 * PB, 1);
  // c3 = conv1x1_c3(cat)    (batched GEMM, K=200)
  launch_gemm(stream, EPI_NONE, c3w, catb, c3out, CHN, HWSZ, 2 * 100, 2 * 100,
              HWSZ, HWSZ, 2 * PB, 2 * PB, BSZ, nullptr, nullptr, 0, 0);
  // out = c3 + xn2 + residual2
  add3_ew<<<gTOT, TB, 0, stream>>>(c3out, xn2, xmid, (float*)d_out, TOTE);
}
